// MsaHmmLayer_58067957841934
// MI455X (gfx1250) — matrix operand order repaired and checked by reading, not compiled, not run on hardware
//
#include <hip/hip_runtime.h>
#include <hip/hip_bf16.h>
#include <math.h>

// ---------------- problem constants ----------------
#define QN     1027           // number of HMM states (2*512+3)
#define QP     1056           // padded: 66 * 16 = 33 * 32
#define NTILES 66             // QP / 16  (N tiles)
#define KTILES 33             // QP / 32  (K chunks of 32)
#define SSYM   26             // alphabet
#define TLEN   512
#define BATCH  256
#define NSPLIT 8              // WGs cooperating on one batch tile
#define NGROUP (BATCH / 16)   // 16 groups
#define MAXNT  9              // max N tiles per WG (66 = 2*9 + 6*8)
#define MAXNC  (MAXNT * 16)   // max columns per WG = 144

typedef _Float16 v16h __attribute__((ext_vector_type(16)));
typedef _Float16 v8h  __attribute__((ext_vector_type(8)));
typedef float    v8f  __attribute__((ext_vector_type(8)));

// ---------------------------------------------------------------------------
// Kernel 1: row softmax of A_logits [QN][QN] -> A_prob fp32 [QN][QN]
// ---------------------------------------------------------------------------
__global__ __launch_bounds__(256) void k_softmax_A(const float* __restrict__ L,
                                                   float* __restrict__ P) {
    __shared__ float buf[256];
    const int r = blockIdx.x;
    const int tid = threadIdx.x;
    float v[5];
    float mx = -3.4e38f;
#pragma unroll
    for (int j = 0; j < 5; ++j) {
        int c = tid + j * 256;
        v[j] = (c < QN) ? L[(size_t)r * QN + c] : -3.4e38f;
        mx = fmaxf(mx, v[j]);
    }
    buf[tid] = mx; __syncthreads();
    for (int s = 128; s > 0; s >>= 1) {
        if (tid < s) buf[tid] = fmaxf(buf[tid], buf[tid + s]);
        __syncthreads();
    }
    mx = buf[0]; __syncthreads();
    float sum = 0.f;
#pragma unroll
    for (int j = 0; j < 5; ++j) {
        int c = tid + j * 256;
        v[j] = __expf(v[j] - mx);
        if (c >= QN) v[j] = 0.f;
        sum += v[j];
    }
    buf[tid] = sum; __syncthreads();
    for (int s = 128; s > 0; s >>= 1) {
        if (tid < s) buf[tid] += buf[tid + s];
        __syncthreads();
    }
    const float inv = 1.f / buf[0];
#pragma unroll
    for (int j = 0; j < 5; ++j) {
        int c = tid + j * 256;
        if (c < QN) P[(size_t)r * QN + c] = v[j] * inv;
    }
}

// ---------------------------------------------------------------------------
// Kernel 2: pack A_prob into WMMA B-operand layout (f16), zero padded.
//   Apk[nt][kc][lane][i]: lane l holds col n = nt*16+(l&15), K = kc*32+16*(l>>4)+i
// ---------------------------------------------------------------------------
__global__ void k_pack_A(const float* __restrict__ P, _Float16* __restrict__ Apk) {
    const int nt = blockIdx.x;
    const int kc = blockIdx.y;
    const int l  = threadIdx.x;          // 0..31
    const int n  = nt * 16 + (l & 15);
    const int h  = l >> 4;
    v16h val;
#pragma unroll
    for (int i = 0; i < 16; ++i) {
        int k = kc * 32 + 16 * h + i;
        float x = (k < QN && n < QN) ? P[(size_t)k * QN + n] : 0.f;
        val[i] = (_Float16)x;
    }
    *(v16h*)(Apk + (((size_t)nt * KTILES + kc) * 32 + l) * 16) = val;
}

// ---------------------------------------------------------------------------
// Kernel 3: row softmax of B_logits -> B_prob [QP][26] (pad rows = 0)
// ---------------------------------------------------------------------------
__global__ __launch_bounds__(32) void k_softmax_B(const float* __restrict__ L,
                                                  float* __restrict__ P) {
    const int r = blockIdx.x;
    const int l = threadIdx.x;
    const bool live = (r < QN) && (l < SSYM);
    float x = live ? L[(size_t)r * SSYM + l] : -3.4e38f;
    float m = x;
#pragma unroll
    for (int off = 16; off > 0; off >>= 1) m = fmaxf(m, __shfl_xor(m, off));
    float e = live ? __expf(x - m) : 0.f;
    float s = e;
#pragma unroll
    for (int off = 16; off > 0; off >>= 1) s += __shfl_xor(s, off);
    if (l < SSYM) P[(size_t)r * SSYM + l] = (r < QN) ? (e / s) : 0.f;
}

// ---------------------------------------------------------------------------
// Kernel 4: softmax of init_logits [QN] -> init_prob [QP] (pad = 0)
// ---------------------------------------------------------------------------
__global__ __launch_bounds__(256) void k_softmax_init(const float* __restrict__ L,
                                                      float* __restrict__ P) {
    __shared__ float buf[256];
    const int tid = threadIdx.x;
    float v[5];
    float mx = -3.4e38f;
#pragma unroll
    for (int j = 0; j < 5; ++j) {
        int c = tid + j * 256;
        v[j] = (c < QN) ? L[c] : -3.4e38f;
        mx = fmaxf(mx, v[j]);
    }
    buf[tid] = mx; __syncthreads();
    for (int s = 128; s > 0; s >>= 1) {
        if (tid < s) buf[tid] = fmaxf(buf[tid], buf[tid + s]);
        __syncthreads();
    }
    mx = buf[0]; __syncthreads();
    float sum = 0.f;
#pragma unroll
    for (int j = 0; j < 5; ++j) {
        int c = tid + j * 256;
        v[j] = __expf(v[j] - mx);
        if (c >= QN) v[j] = 0.f;
        sum += v[j];
    }
    buf[tid] = sum; __syncthreads();
    for (int s = 128; s > 0; s >>= 1) {
        if (tid < s) buf[tid] += buf[tid + s];
        __syncthreads();
    }
    const float inv = 1.f / buf[0];
#pragma unroll
    for (int j = 0; j < 5; ++j) {
        int c = tid + j * 256;
        if (c < QP) P[c] = (c < QN) ? v[j] * inv : 0.f;
    }
}

// ---------------------------------------------------------------------------
// Kernel 5: zero the inter-WG barrier counters (must run every launch).
// ---------------------------------------------------------------------------
__global__ void k_zero(unsigned* __restrict__ ctr, int n) {
    int i = blockIdx.x * blockDim.x + threadIdx.x;
    if (i < n) ctr[i] = 0u;
}

// ---------------------------------------------------------------------------
// helpers: async global->LDS copy (CDNA5, ASYNCcnt) and group split-barrier
// ---------------------------------------------------------------------------
__device__ __forceinline__ void async_copy_b128(void* lds_dst, const void* gsrc) {
    // flat->LDS addrspace: low 32 bits of the flat shared address are the
    // LDS byte address (ISA 10.2 aperture mapping / LLVM addrspacecast).
    unsigned l = (unsigned)(unsigned long long)lds_dst;
    asm volatile("global_load_async_to_lds_b128 %0, %1, off"
                 :: "v"(l), "v"(gsrc) : "memory");
}
__device__ __forceinline__ void wait_async0() {
    asm volatile("s_wait_asynccnt 0x0" ::: "memory");
}

__device__ __forceinline__ void group_barrier(unsigned* __restrict__ c,
                                              unsigned target, int tid) {
    __builtin_amdgcn_fence(__ATOMIC_RELEASE, "agent");   // flush this thread's stores
    __syncthreads();                                     // whole WG flushed
    if (tid == 0) {
        atomicAdd(c, 1u);
        while (__hip_atomic_load(c, __ATOMIC_RELAXED, __HIP_MEMORY_SCOPE_AGENT)
               < target)
            __builtin_amdgcn_s_sleep(2);
    }
    __syncthreads();
    __builtin_amdgcn_fence(__ATOMIC_ACQUIRE, "agent");   // invalidate stale lines
}

// ---------------------------------------------------------------------------
// Kernel 6: persistent forward recursion, N-split across a group of 8 WGs.
// grid = 128 WGs: group g = blockIdx>>3 owns batch rows [16g,16g+16);
// split p = blockIdx&7 owns 8-9 N-tiles. alpha exchanged per step in f16 via
// global memory, gathered with global_load_async_to_lds_b128.
// ---------------------------------------------------------------------------
__global__ __launch_bounds__(256) void k_forward(const int* __restrict__ seq,
                                                 const _Float16* __restrict__ Apk,
                                                 const float* __restrict__ Bp,
                                                 const float* __restrict__ initp,
                                                 _Float16* __restrict__ galpha,
                                                 float* __restrict__ gpart,
                                                 unsigned* __restrict__ ctr,
                                                 float* __restrict__ out) {
    extern __shared__ char smem[];
    _Float16* a_lds  = (_Float16*)smem;                              // [16][QP] f16
    float*    f_part = (float*)(smem + (size_t)16 * QP * 2);         // [16][MAXNC]
    float*    Bsl    = f_part + 16 * MAXNC;                          // [MAXNC][26]
    float*    red    = Bsl + MAXNC * SSYM;                           // [256]
    float*    srec   = red + 256;                                    // [16]
    float*    ll     = srec + 16;                                    // [16]
    int*      sym    = (int*)(ll + 16);                              // [16]

    const int tid  = threadIdx.x;
    const int lane = tid & 31;
    const int wave = tid >> 5;
    const int g    = blockIdx.x >> 3;        // batch-tile group
    const int p    = blockIdx.x & (NSPLIT - 1);
    const int b0   = g * 16;
    const int m    = lane & 15;
    const int h    = lane >> 4;

    const int ntbase = p * 8 + (p < 2 ? p : 2);   // contiguous tile range
    const int cnt    = (p < 2) ? 9 : 8;
    const int ncol0  = ntbase * 16;
    const int ncols  = cnt * 16;

    _Float16* myalpha = galpha + (size_t)g * 16 * QP;
    float*    mypart  = gpart + (size_t)g * NSPLIT * 16;
    unsigned* c1      = ctr + g * 2;
    unsigned* c2      = ctr + g * 2 + 1;

    // Stage this WG's emission-table slice into LDS.
    for (int i = tid; i < ncols * SSYM; i += 256)
        Bsl[i] = Bp[(size_t)(ncol0 + i / SSYM) * SSYM + (i % SSYM)];
    if (tid < 16) { ll[tid] = 0.f; sym[tid] = seq[(size_t)(b0 + tid) * TLEN]; }
    __syncthreads();

    unsigned st = 1;                 // step index for monotonic barrier targets

    // ---------------- t = 0 : f = init * E[:,0] ----------------
    for (int i = tid; i < 16 * ncols; i += 256) {
        int r = i / ncols, ln = i % ncols;
        f_part[i] = initp[ncol0 + ln] * Bsl[ln * SSYM + sym[r]];
    }
    __syncthreads();

    for (int t = 0; t < TLEN; ++t) {
        // ---- partial row sums of f_part -> global ----
        {
            const int r = tid >> 4, c = tid & 15;
            float ps = 0.f;
            for (int j = 0; j < cnt; ++j) ps += f_part[r * ncols + c + 16 * j];
            red[tid] = ps;
        }
        __syncthreads();
        if (tid < 16) {
            float ps = 0.f;
#pragma unroll
            for (int j = 0; j < 16; ++j) ps += red[tid * 16 + j];
            mypart[p * 16 + tid] = ps;
        }
        group_barrier(c1, NSPLIT * st, tid);

        // ---- exact scale, loglik, publish own scaled f16 alpha columns ----
        if (tid < 16) {
            float s = 0.f;
#pragma unroll
            for (int q = 0; q < NSPLIT; ++q) s += mypart[q * 16 + tid];
            ll[tid] += logf(s);
            srec[tid] = 1.0f / s;
        }
        __syncthreads();
        for (int i = tid; i < 16 * ncols; i += 256) {
            int r = i / ncols, ln = i % ncols;
            myalpha[(size_t)r * QP + ncol0 + ln] = (_Float16)(f_part[i] * srec[r]);
        }
        group_barrier(c2, NSPLIT * st, tid);
        ++st;
        if (t == TLEN - 1) break;

        // ---- gather full alpha into LDS via async global->LDS copies ----
        {
            const char* src = (const char*)myalpha;
            char*       dst = (char*)a_lds;
            for (int i = tid; i < (16 * QP * 2) / 16; i += 256)
                async_copy_b128(dst + (size_t)i * 16, src + (size_t)i * 16);
            wait_async0();
        }
        if (tid < 16) sym[tid] = seq[(size_t)(b0 + tid) * TLEN + (t + 1)];
        __syncthreads();

        // ---- WMMA: f_part = (alpha @ A)[:, mycols] * e_t ----
        for (int j = wave; j < cnt; j += 8) {
            const int ntg = ntbase + j;
            v8f acc = {};
            const _Float16* bp = Apk + (((size_t)ntg * KTILES) * 32 + lane) * 16;
#pragma unroll 3
            for (int kc = 0; kc < KTILES; ++kc) {
                const _Float16* arow = a_lds + m * QP + kc * 32 + 16 * h;
                v8h lo = *(const v8h*)(arow);
                v8h hi = *(const v8h*)(arow + 8);
                v16h a = __builtin_shufflevector(lo, hi, 0, 1, 2, 3, 4, 5, 6, 7,
                                                 8, 9, 10, 11, 12, 13, 14, 15);
                v16h b = *(const v16h*)bp;
                bp += 32 * 16;
                __builtin_prefetch(bp, 0, 1);
                acc = __builtin_amdgcn_wmma_f32_16x16x32_f16(
                        false, a, false, b, (short)0, acc, false, false);
            }
            const int nl = j * 16 + m;
            float ev[8];
#pragma unroll
            for (int v = 0; v < 8; ++v) ev[v] = Bsl[nl * SSYM + sym[v + 8 * h]];
#pragma unroll
            for (int v = 0; v < 8; ++v)
                f_part[(v + 8 * h) * ncols + nl] = acc[v] * ev[v];
        }
        __syncthreads();
    }

    if (p == 0 && tid < 16) out[b0 + tid] = ll[tid];
}

// ---------------------------------------------------------------------------
// host-side launcher
// ---------------------------------------------------------------------------
static inline size_t align_up(size_t x, size_t a) { return (x + a - 1) & ~(a - 1); }

extern "C" void kernel_launch(void* const* d_in, const int* in_sizes, int n_in,
                              void* d_out, int out_size, void* d_ws, size_t ws_size,
                              hipStream_t stream) {
    (void)in_sizes; (void)n_in; (void)out_size; (void)ws_size;
    const int*   seq  = (const int*)d_in[0];           // [256,512] int32
    const float* Alog = (const float*)d_in[1];         // [Q,Q]
    const float* Blog = (const float*)d_in[2];         // [Q,S]
    const float* Ilog = (const float*)d_in[3];         // [Q]
    float*       out  = (float*)d_out;                 // [256]

    char* ws = (char*)d_ws;
    size_t off = 0;
    float* A_prob = (float*)(ws + off);
    off = align_up(off + (size_t)QN * QN * sizeof(float), 256);
    _Float16* Apk = (_Float16*)(ws + off);
    off = align_up(off + (size_t)NTILES * KTILES * 32 * 16 * sizeof(_Float16), 256);
    float* B_prob = (float*)(ws + off);
    off = align_up(off + (size_t)QP * SSYM * sizeof(float), 256);
    float* init_prob = (float*)(ws + off);
    off = align_up(off + (size_t)QP * sizeof(float), 256);
    _Float16* galpha = (_Float16*)(ws + off);
    off = align_up(off + (size_t)NGROUP * 16 * QP * sizeof(_Float16), 256);
    float* gpart = (float*)(ws + off);
    off = align_up(off + (size_t)NGROUP * NSPLIT * 16 * sizeof(float), 256);
    unsigned* ctr = (unsigned*)(ws + off);
    off = align_up(off + (size_t)NGROUP * 2 * sizeof(unsigned), 256);

    k_zero<<<1, 64, 0, stream>>>(ctr, NGROUP * 2);
    k_softmax_A<<<QN, 256, 0, stream>>>(Alog, A_prob);
    k_pack_A<<<dim3(NTILES, KTILES), 32, 0, stream>>>(A_prob, Apk);
    k_softmax_B<<<QP, 32, 0, stream>>>(Blog, B_prob);
    k_softmax_init<<<1, 256, 0, stream>>>(Ilog, init_prob);

    const size_t lds_bytes = (size_t)16 * QP * 2        // alpha f16
                           + (size_t)16 * MAXNC * 4     // f_part
                           + (size_t)MAXNC * SSYM * 4   // B slice
                           + 256 * 4 + 16 * 4 + 16 * 4 + 16 * 4;
    k_forward<<<NGROUP * NSPLIT, 256, lds_bytes, stream>>>(
        seq, Apk, B_prob, init_prob, galpha, gpart, ctr, out);
}
